// Energyator_53979148976569
// MI455X (gfx1250) — matrix operand order repaired and checked by reading, not compiled, not run on hardware
//
#include <hip/hip_runtime.h>
#include <hip/hip_bf16.h>
#include <math.h>

// Problem constants (from reference): S=8 species, B=65536 batch, F=U=256.
#define NSPEC 8
#define NBATCH 65536
#define NFEAT 256
#define NUNIT 256

#define BT   128           // batch rows per block (halves L2 weight re-staging vs 64)
#define NH   128           // N-half width staged per weight tile
#define LDX  (NFEAT + 8)   // activation LDS row stride (elements), breaks 64-bank periodicity
#define LDW  (NUNIT + 8)   // transposed-weight LDS row stride (elements)

typedef __attribute__((ext_vector_type(16))) __bf16 v16bf;
typedef __attribute__((ext_vector_type(8)))  float  v8f;

union Frag {
    unsigned int u32[8];
    v16bf        v;
};

__device__ __forceinline__ unsigned short f2bf(float f) {
    unsigned int u = __builtin_bit_cast(unsigned int, f);
    unsigned int r = u + 0x7FFFu + ((u >> 16) & 1u);  // round-to-nearest-even
    return (unsigned short)(r >> 16);
}
__device__ __forceinline__ float bf2f(unsigned short h) {
    unsigned int u = ((unsigned int)h) << 16;
    return __builtin_bit_cast(float, u);
}

__global__ __launch_bounds__(256, 1)
void energyator_mlp_wmma(const float* __restrict__ gB,
                         const float* __restrict__ gW1, const float* __restrict__ gb1,
                         const float* __restrict__ gW2, const float* __restrict__ gb2,
                         const float* __restrict__ gW3, const float* __restrict__ gb3,
                         const float* __restrict__ gW4, const float* __restrict__ gb4,
                         float* __restrict__ out) {
    __shared__ unsigned short sX[BT][LDX];   // activation ping (128x256 bf16)
    __shared__ unsigned short sY[BT][LDX];   // activation pong
    __shared__ unsigned short sWt[NH][LDW];  // transposed bf16 weight half: sWt[n][k]

    const int tid     = threadIdx.x;
    const int lane    = tid & 31;
    const int wave    = tid >> 5;      // 0..7
    const int half    = lane >> 4;     // 0: lanes 0-15, 1: lanes 16-31
    const int rsel    = lane & 15;
    const int rowBase = blockIdx.x * BT;
    const int m0      = wave * 16;     // each wave owns one 16-row strip of the 128-row tile

    float accA = 0.0f;  // per-row sum over species of final-layer output (threads 0..BT-1)

    for (int s = 0; s < NSPEC; ++s) {
        // ---- stage input tile: fp32 global -> bf16 LDS (128x256) ----
        const float* xg = gB + ((size_t)s * NBATCH + rowBase) * NFEAT;
        for (int i = tid; i < BT * (NFEAT / 4); i += 256) {
            int r  = i >> 6;          // / (NFEAT/4)
            int c4 = i & 63;
            float4 v = ((const float4*)(xg + (size_t)r * NFEAT))[c4];
            int c = c4 * 4;
            sX[r][c + 0] = f2bf(v.x);
            sX[r][c + 1] = f2bf(v.y);
            sX[r][c + 2] = f2bf(v.z);
            sX[r][c + 3] = f2bf(v.w);
        }
        __syncthreads();

        unsigned short (*Xcur)[LDX]  = sX;
        unsigned short (*Ynext)[LDX] = sY;

        const float* Wgs[3] = { gW1 + (size_t)s * NFEAT * NUNIT,
                                gW2 + (size_t)s * NUNIT * NUNIT,
                                gW3 + (size_t)s * NUNIT * NUNIT };
        const float* bgs[3] = { gb1 + s * NUNIT, gb2 + s * NUNIT, gb3 + s * NUNIT };

        for (int L = 0; L < 3; ++L) {
            const float* Wg = Wgs[L];
            const float* bg = bgs[L];
            for (int nh = 0; nh < 2; ++nh) {
                const int n0h = nh * NH;
                // ---- stage weight half, transposed, fp32 -> bf16: sWt[n][k] = W[k][n0h+n] ----
                for (int i = tid; i < NUNIT * (NH / 4); i += 256) {
                    int k  = i >> 5;      // / (NH/4)
                    int c4 = i & 31;
                    float4 v = ((const float4*)(Wg + (size_t)k * NUNIT + n0h))[c4];
                    int c = c4 * 4;
                    sWt[c + 0][k] = f2bf(v.x);
                    sWt[c + 1][k] = f2bf(v.y);
                    sWt[c + 2][k] = f2bf(v.z);
                    sWt[c + 3][k] = f2bf(v.w);
                }
                __syncthreads();

                // Hint-prefetch next layer's weights into GL2 while we compute.
                if (L + 1 < 3)
                    __builtin_prefetch((const char*)Wgs[L + 1] + (size_t)tid * 256, 0, 1);

                // ---- 8 waves x 8 column tiles cover the 128x128 output half ----
                // A-strip (rows m0..m0+15, all K) is loop-invariant across the 8 tiles.
                const unsigned int* arow = (const unsigned int*)Xcur[m0 + rsel];
                for (int t = 0; t < 8; ++t) {
                    const int nW   = t * 16;       // column base within the staged half
                    const int nAbs = n0h + nW;     // absolute output column base

                    v8f acc = {};
                    const unsigned int* brow = (const unsigned int*)sWt[nW + rsel];

                    for (int kb = 0; kb < NUNIT; kb += 32) {
                        Frag A, Bf;
                        // A 16x32 bf16 fragment (ISA 7.12.2 layout):
                        //   lanes 0-15 row M=rsel: V0..3 K=0..7, V4..7 K=16..23
                        //   lanes 16-31:           V0..3 K=8..15, V4..7 K=24..31
                        #pragma unroll
                        for (int v = 0; v < 8; ++v) {
                            int kk = (v < 4) ? (kb + half * 8 + 2 * v)
                                             : (kb + 16 + half * 8 + 2 * (v - 4));
                            A.u32[v] = arow[kk >> 1];
                        }
                        // B 32x16 bf16 fragment: lane column N=rsel,
                        //   lanes 0-15: V0..3 K=0..7, V4..7 K=16..23; lanes 16-31: V0..3 K=8..15, V4..7 K=24..31
                        #pragma unroll
                        for (int v = 0; v < 8; ++v) {
                            int kk = (v < 4) ? (kb + half * 8 + 2 * v)
                                             : (kb + 16 + half * 8 + 2 * (v - 4));
                            Bf.u32[v] = brow[kk >> 1];
                        }
                        acc = __builtin_amdgcn_wmma_f32_16x16x32_bf16(
                                  false, A.v, false, Bf.v, (short)0, acc, false, false);
                    }

                    // Epilogue: bias + LeakyReLU(0.1), bf16 back to ping-pong buffer.
                    // C/D layout: VGPR v, lanes 0-15 -> M=m0+v; lanes 16-31 -> M=m0+v+8
                    const int n = nAbs + rsel;
                    const float bias = bg[n];
                    #pragma unroll
                    for (int v = 0; v < 8; ++v) {
                        int m = m0 + v + half * 8;
                        float val = acc[v] + bias;
                        val = (val >= 0.0f) ? val : 0.1f * val;
                        Ynext[m][n] = f2bf(val);
                    }
                }
                __syncthreads();  // half done before restaging sWt / consuming Ynext
            }
            // swap ping-pong
            unsigned short (*tmp)[LDX] = Xcur; Xcur = Ynext; Ynext = tmp;
        }

        // ---- layer 4: per-row 256-dot with W4 (S,U,1) + b4, accumulate over species ----
        if (tid < BT) {
            const float* w4 = gW4 + (size_t)s * NUNIT;
            float dot = 0.0f;
            #pragma unroll 4
            for (int k = 0; k < NUNIT; ++k)
                dot += bf2f(Xcur[tid][k]) * w4[k];
            accA += dot + gb4[s];
        }
        __syncthreads();  // reads of final activations done before next species restages
    }

    if (tid < BT)
        out[rowBase + tid] = 1.0f / (1.0f + expf(-accA));
}

extern "C" void kernel_launch(void* const* d_in, const int* in_sizes, int n_in,
                              void* d_out, int out_size, void* d_ws, size_t ws_size,
                              hipStream_t stream) {
    (void)in_sizes; (void)n_in; (void)out_size; (void)d_ws; (void)ws_size;
    const float* b  = (const float*)d_in[0];
    const float* W1 = (const float*)d_in[1];
    const float* b1 = (const float*)d_in[2];
    const float* W2 = (const float*)d_in[3];
    const float* b2 = (const float*)d_in[4];
    const float* W3 = (const float*)d_in[5];
    const float* b3 = (const float*)d_in[6];
    const float* W4 = (const float*)d_in[7];
    const float* b4 = (const float*)d_in[8];
    float* out = (float*)d_out;

    dim3 grid(NBATCH / BT);   // 512 blocks
    dim3 block(256);          // 8 wave32 waves
    energyator_mlp_wmma<<<grid, block, 0, stream>>>(b, W1, b1, W2, b2, W3, b3, W4, b4, out);
}
